// TernConv2d_2173253452085
// MI455X (gfx1250) — matrix operand order repaired and checked by reading, not compiled, not run on hardware
//
#include <hip/hip_runtime.h>
#include <hip/hip_bf16.h>

// ---------------- problem constants ----------------
#define N_BATCH 64
#define C_IN    128
#define H_IN    64
#define W_IN    64
#define C_OUT   128
#define H_OUT   62
#define W_OUT   62
#define P_TOT   (H_OUT * W_OUT)           // 3844 spatial outputs per image
#define M_TILE  128                       // spatial tile per block
#define W_ELEMS (C_OUT * C_IN * 9)        // 147456
#define G_ELEMS (16 * C_IN * H_IN * W_IN) // elements per batch-group: 8388608
#define HW_IN   (H_IN * W_IN)             // 4096

typedef __attribute__((ext_vector_type(16))) _Float16 v16h;
typedef __attribute__((ext_vector_type(8)))  _Float16 v8h;
typedef __attribute__((ext_vector_type(8)))  float    v8f;
typedef int v4i_t __attribute__((vector_size(16)));   // matches builtin param type

#define AS1 __attribute__((address_space(1)))
#define AS3 __attribute__((address_space(3)))

#if __has_builtin(__builtin_amdgcn_global_load_async_to_lds_b128) && \
    __has_builtin(__builtin_amdgcn_s_wait_asynccnt)
#define HAVE_ASYNC_LDS 1
#endif

// ---------------- 1: zero per-group absmax slots ----------------
__global__ void init_ws_kernel(unsigned* __restrict__ gmax) {
    if (threadIdx.x < 4) gmax[threadIdx.x] = 0u;
}

// ---------------- 2: weight mean (single block, deterministic) ----------------
__global__ __launch_bounds__(256) void wmean_kernel(const float* __restrict__ w,
                                                    float* __restrict__ alpha_out) {
    __shared__ float sm[256];
    float s = 0.0f;
    for (int i = threadIdx.x; i < W_ELEMS; i += 256) s += w[i];
    sm[threadIdx.x] = s;
    __syncthreads();
    for (int o = 128; o > 0; o >>= 1) {
        if (threadIdx.x < (unsigned)o) sm[threadIdx.x] += sm[threadIdx.x + o];
        __syncthreads();
    }
    if (threadIdx.x == 0) alpha_out[0] = sm[0] / (float)W_ELEMS;
}

// ---------------- 3: ternarize + prepack B ----------------
// Packed layout: Bw[tap][co][ci]  (32 KB per tap -> straight linear LDS copy in conv)
__global__ __launch_bounds__(256) void ternarize_kernel(const float* __restrict__ w,
                                                        const float* __restrict__ alpha_p,
                                                        _Float16* __restrict__ Bw) {
    int e = blockIdx.x * 256 + threadIdx.x;
    if (e >= W_ELEMS) return;
    float alpha = alpha_p[0];
    int co  = e / (C_IN * 9);
    int r   = e - co * (C_IN * 9);
    int ci  = r / 9;
    int tap = r - ci * 9;                 // kh*3+kw
    float wc = w[e] - alpha;
    float t  = (wc > 0.05f) ? 1.0f : ((wc < -0.05f) ? -1.0f : 0.0f);
    Bw[(size_t)tap * (C_OUT * C_IN) + (size_t)co * C_IN + ci] = (_Float16)t;
}

// ---------------- 4: per-group abs-max (bitwise atomicMax == exact & deterministic) ----
__global__ __launch_bounds__(256) void absmax_kernel(const float* __restrict__ x,
                                                     unsigned* __restrict__ gmax) {
    __shared__ unsigned sm[256];
    const int g = blockIdx.y;
    const float* xg = x + (size_t)g * G_ELEMS;
    unsigned m = 0u;
    for (long i = (long)blockIdx.x * 256 + threadIdx.x; i < G_ELEMS; i += (long)gridDim.x * 256) {
        unsigned b = __float_as_uint(fabsf(xg[i]));   // non-negative float: uint order == float order
        m = (b > m) ? b : m;
    }
    sm[threadIdx.x] = m;
    __syncthreads();
    for (int o = 128; o > 0; o >>= 1) {
        if (threadIdx.x < (unsigned)o) {
            unsigned b = sm[threadIdx.x + o];
            if (b > sm[threadIdx.x]) sm[threadIdx.x] = b;
        }
        __syncthreads();
    }
    if (threadIdx.x == 0) atomicMax(&gmax[g], sm[0]);
}

// ---- B tile copy: async global->LDS (ASYNCcnt path) with synchronous fallback ----
__device__ __forceinline__ void copy_b_tile(const _Float16* __restrict__ gsrc,
                                            _Float16* lds_dst, int tid) {
#ifdef HAVE_ASYNC_LDS
    AS1 v4i_t* g = (AS1 v4i_t*)(v4i_t*)const_cast<_Float16*>(gsrc);
    AS3 v4i_t* l = (AS3 v4i_t*)(v4i_t*)lds_dst;
    #pragma unroll
    for (int j = 0; j < 8; ++j) {
        __builtin_amdgcn_global_load_async_to_lds_b128(
            g + (tid + 256 * j), l + (tid + 256 * j), 0, 0);
    }
#else
    const v8h* s = (const v8h*)gsrc;
    v8h* d = (v8h*)lds_dst;
    #pragma unroll
    for (int j = 0; j < 8; ++j) d[tid + 256 * j] = s[tid + 256 * j];
#endif
}

// ---------------- 5: implicit-GEMM conv via v_wmma_f32_16x16x32_f16 ----------------
// Block: 256 threads (8 waves). Tile: 128 spatial x 128 Cout for one batch image.
// Tap loop (9 iters): A = 128 m x 128 ci (32KB, gather+quantize) staged synchronously;
// B = 128 co x 128 ci (32KB) double-buffered via GLOBAL_LOAD_ASYNC_TO_LDS_B128 so the
// next tap's weights stream in during A staging + compute. 32 WMMAs/wave per barrier.
// Wave w owns spatial rows [w*16, w*16+16) and all 128 Cout (8 accumulators).
__global__ __launch_bounds__(256) void conv_wmma_kernel(const float* __restrict__ x,
                                                        const _Float16* __restrict__ Bw,
                                                        const unsigned* __restrict__ gbits,
                                                        float* __restrict__ out) {
    __shared__ __align__(16) _Float16 Ash[M_TILE][C_IN];      // 32 KB
    __shared__ __align__(16) _Float16 Bsh[2][C_OUT][C_IN];    // 2 x 32 KB (double buffer)

    const int tid  = threadIdx.x;
    const int lane = tid & 31;
    const int wave = tid >> 5;
    const int half = lane >> 4;     // 0: lanes 0-15, 1: lanes 16-31
    const int l16  = lane & 15;

    const int n  = blockIdx.y;                  // batch image
    const int p0 = blockIdx.x * M_TILE;         // first spatial output of tile
    const float gamma = __uint_as_float(gbits[n >> 4]);
    const float scale = 128.0f / (gamma + 1e-3f);
    const float lo = -128.0f + 1e-3f, hi = 128.0f - 1e-3f;

    const float* xn = x + (size_t)n * C_IN * HW_IN;

    v8f acc[8] = {v8f{}, v8f{}, v8f{}, v8f{}, v8f{}, v8f{}, v8f{}, v8f{}};

    // A loader mapping: thread -> (m = tid&127, ci block = (tid>>7)*64)
    // Per fixed ci, a wave's 32 lanes read 32 consecutive spatial positions -> coalesced.
    const int la_m  = tid & 127;
    const int la_c0 = (tid >> 7) * 64;
    int pA = p0 + la_m; if (pA >= P_TOT) pA = P_TOT - 1;   // clamp (stores guarded)
    const int ohA = pA / W_OUT;
    const int owA = pA - ohA * W_OUT;
    const float* xA = xn + (size_t)la_c0 * HW_IN + (size_t)ohA * W_IN + owA;

    // prefetch tap 0 weights into buffer 0
    copy_b_tile(Bw, &Bsh[0][0][0], tid);

    for (int tap = 0; tap < 9; ++tap) {
        const int kh = tap / 3;
        const int kw = tap - kh * 3;
        const int buf = tap & 1;

        // ---- stage A: gather + quantize 128 m x 128 ci activations ----
        {
            const float* src = xA + kh * W_IN + kw;
            #pragma unroll 8
            for (int j = 0; j < 64; ++j) {
                float v = src[(size_t)j * HW_IN];
                v = fminf(fmaxf(v * scale, lo), hi);
                Ash[la_m][la_c0 + j] = (_Float16)v;
            }
        }
        // ---- kick off next tap's B copy into the other buffer ----
        if (tap < 8)
            copy_b_tile(Bw + (size_t)(tap + 1) * (C_OUT * C_IN),
                        &Bsh[buf ^ 1][0][0], tid);

#ifdef HAVE_ASYNC_LDS
        // async loads complete in order: <=8 outstanding => current buffer landed
        if (tap < 8) __builtin_amdgcn_s_wait_asynccnt(8);
        else         __builtin_amdgcn_s_wait_asynccnt(0);
#endif
        __syncthreads();

        // ---- compute: 4 ci-sub-slices (K=32 each) x 8 Cout fragments ----
        #pragma unroll
        for (int sub = 0; sub < 4; ++sub) {
            const int ci0 = sub * 32;
            // A fragment (ISA layout: lane half h holds k = h*8..+7 and 16+h*8..+7)
            const v8h* arow = (const v8h*)&Ash[wave * 16 + l16][ci0];
            v8h aLo = arow[half];
            v8h aHi = arow[2 + half];
            v16h afrag = __builtin_shufflevector(aLo, aHi,
                            0,1,2,3,4,5,6,7,8,9,10,11,12,13,14,15);
            #pragma unroll
            for (int f = 0; f < 8; ++f) {
                // B fragment (lane half h holds k = h*8..+7 and 16+h*8..+7, same as A)
                const v8h* brow = (const v8h*)&Bsh[buf][f * 16 + l16][ci0];
                v8h bLo = brow[half];
                v8h bHi = brow[2 + half];
                v16h bfrag = __builtin_shufflevector(bLo, bHi,
                                0,1,2,3,4,5,6,7,8,9,10,11,12,13,14,15);
                acc[f] = __builtin_amdgcn_wmma_f32_16x16x32_f16(
                            false, afrag, false, bfrag, (short)0, acc[f], false, false);
            }
        }
        __syncthreads();   // protects Ash (next A stage) and Bsh[buf] (next async copy)
    }

    // ---- store D: VGPR r holds (M = r + 8*half, N = l16) ----
    #pragma unroll
    for (int f = 0; f < 8; ++f) {
        const int co = f * 16 + l16;
        float* obase = out + ((size_t)n * C_OUT + co) * P_TOT;
        #pragma unroll
        for (int r = 0; r < 8; ++r) {
            int p = p0 + wave * 16 + half * 8 + r;
            if (p < P_TOT) obase[p] = acc[f][r];
        }
    }
}

// ---------------- launch ----------------
extern "C" void kernel_launch(void* const* d_in, const int* in_sizes, int n_in,
                              void* d_out, int out_size, void* d_ws, size_t ws_size,
                              hipStream_t stream) {
    const float* x = (const float*)d_in[0];   // (64,128,64,64)
    const float* w = (const float*)d_in[1];   // (128,128,3,3)
    float* out = (float*)d_out;               // (64,128,62,62)

    unsigned*  ws_u  = (unsigned*)d_ws;               // [0..3] group absmax bits
    float*     ws_f  = (float*)d_ws;                  // [4] alpha
    _Float16*  Bw    = (_Float16*)((char*)d_ws + 64); // packed ternary weights: 9*128*128 f16

    init_ws_kernel<<<dim3(1), dim3(32), 0, stream>>>(ws_u);
    wmean_kernel<<<dim3(1), dim3(256), 0, stream>>>(w, ws_f + 4);
    ternarize_kernel<<<dim3((W_ELEMS + 255) / 256), dim3(256), 0, stream>>>(w, ws_f + 4, Bw);
    absmax_kernel<<<dim3(256, 4), dim3(256), 0, stream>>>(x, ws_u);
    conv_wmma_kernel<<<dim3((P_TOT + M_TILE - 1) / M_TILE, N_BATCH), dim3(256), 0, stream>>>(
        x, Bw, ws_u, out);
}
